// RelPartialLearnableMultiHeadAttn_55439437857178
// MI455X (gfx1250) — compile-verified, matrix operand order checked by reading, not run on hardware
//
#include <hip/hip_runtime.h>
#include <hip/hip_bf16.h>
#include <math.h>

// ---------------------------------------------------------------------------
// Transformer-XL relative multi-head attention for MI455X (gfx1250, wave32).
// All heavy math through v_wmma_f32_16x16x32_bf16 (f32 accumulate).
// ---------------------------------------------------------------------------

typedef __attribute__((ext_vector_type(16))) __bf16          v16bf;
typedef __attribute__((ext_vector_type(8)))  float           v8f;
typedef __attribute__((ext_vector_type(8)))  unsigned short  us8;
typedef __attribute__((ext_vector_type(16))) unsigned short  us16;

#define QLEN   1024
#define KLEN   2048
#define BSZ    2
#define NHEAD  16
#define DHEAD  64
#define DMODEL 1024
#define SCALE  0.125f
#define NEG_BIG (-1.0e30f)

static __device__ __forceinline__ unsigned short f2bf(float f) {
    unsigned u = __builtin_bit_cast(unsigned, f);
    unsigned r = (u + 0x7FFFu + ((u >> 16) & 1u)) >> 16;
    return (unsigned short)r;
}

// Build a 16-element bf16 WMMA operand fragment for this lane.
// Layout per ISA: elements 0..7 at k = kbase + 8*half, 8..15 at kbase+16+8*half.
static __device__ __forceinline__ v16bf load_frag(const unsigned short* rowptr,
                                                  int kbase, int half) {
    const us8* p0 = reinterpret_cast<const us8*>(rowptr + kbase + 8 * half);
    const us8* p1 = reinterpret_cast<const us8*>(rowptr + kbase + 16 + 8 * half);
    us8 lo = *p0;
    us8 hi = *p1;
    us16 u = __builtin_shufflevector(lo, hi, 0, 1, 2, 3, 4, 5, 6, 7,
                                     8, 9, 10, 11, 12, 13, 14, 15);
    return __builtin_bit_cast(v16bf, u);
}

static __device__ __forceinline__ v8f wmma_bf16(v16bf a, v16bf b, v8f c) {
    return __builtin_amdgcn_wmma_f32_16x16x32_bf16(false, a, false, b,
                                                   (short)0, c, false, false);
}

// ---------------------------------------------------------------------------
// Elementwise helpers
// ---------------------------------------------------------------------------
__global__ void k_f2bf(const float* __restrict__ s, unsigned short* __restrict__ d, int n) {
    for (int i = blockIdx.x * blockDim.x + threadIdx.x; i < n; i += gridDim.x * blockDim.x)
        d[i] = f2bf(s[i]);
}

// cat = concat(mems, w) along sequence, converted to bf16. Row = j*BSZ + b.
__global__ void k_build_cat(const float* __restrict__ w, const float* __restrict__ mems,
                            unsigned short* __restrict__ catb) {
    const int n = KLEN * BSZ * DMODEL;
    const int half = QLEN * BSZ * DMODEL;   // mems part size
    for (int i = blockIdx.x * blockDim.x + threadIdx.x; i < n; i += gridDim.x * blockDim.x) {
        float v = (i < half) ? mems[i] : w[i - half];
        catb[i] = f2bf(v);
    }
}

// q (+bq) with the two biases added, stored [b][n][i][d] bf16
__global__ void k_make_q(const float* __restrict__ qf, const float* __restrict__ bq,
                         const float* __restrict__ rwb, const float* __restrict__ rrb,
                         unsigned short* __restrict__ qw, unsigned short* __restrict__ qr) {
    const int n = BSZ * NHEAD * QLEN * DHEAD;
    for (int idx = blockIdx.x * blockDim.x + threadIdx.x; idx < n; idx += gridDim.x * blockDim.x) {
        int d = idx & 63, i = (idx >> 6) & 1023, h = (idx >> 16) & 15, b = (idx >> 20) & 1;
        int col = (h << 6) + d;
        float v = qf[((i * BSZ + b) << 10) + col] + bq[col];
        qw[idx] = f2bf(v + rwb[col]);
        qr[idx] = f2bf(v + rrb[col]);
    }
}

// k stored [b][n][j][d] bf16
__global__ void k_make_k(const float* __restrict__ kf, const float* __restrict__ bk,
                         unsigned short* __restrict__ kb) {
    const int n = BSZ * NHEAD * KLEN * DHEAD;
    for (int idx = blockIdx.x * blockDim.x + threadIdx.x; idx < n; idx += gridDim.x * blockDim.x) {
        int d = idx & 63, j = (idx >> 6) & 2047, h = (idx >> 17) & 15, b = (idx >> 21) & 1;
        int col = (h << 6) + d;
        kb[idx] = f2bf(kf[((j * BSZ + b) << 10) + col] + bk[col]);
    }
}

// v stored transposed [b][n][d][j] bf16 (PV B-fragment reads contiguous j-runs)
__global__ void k_make_vt(const float* __restrict__ vf, const float* __restrict__ bv,
                          unsigned short* __restrict__ vtb) {
    const int n = BSZ * NHEAD * DHEAD * KLEN;
    for (int idx = blockIdx.x * blockDim.x + threadIdx.x; idx < n; idx += gridDim.x * blockDim.x) {
        int j = idx & 2047, d = (idx >> 11) & 63, h = (idx >> 17) & 15, b = (idx >> 21) & 1;
        int col = (h << 6) + d;
        vtb[idx] = f2bf(vf[((j * BSZ + b) << 10) + col] + bv[col]);
    }
}

// rk stored [n][p][d] bf16
__global__ void k_make_rk(const float* __restrict__ rkf, const float* __restrict__ br,
                          unsigned short* __restrict__ rkb) {
    const int n = NHEAD * KLEN * DHEAD;
    for (int idx = blockIdx.x * blockDim.x + threadIdx.x; idx < n; idx += gridDim.x * blockDim.x) {
        int d = idx & 63, p = (idx >> 6) & 2047, h = (idx >> 17) & 15;
        int col = (h << 6) + d;
        rkb[idx] = f2bf(rkf[(p << 10) + col] + br[col]);
    }
}

// ---------------------------------------------------------------------------
// Tiled bf16 WMMA GEMM: C[M,N](f32) = A[M,K](bf16,row-major) @ B[K,N](bf16)
// Block = 128 threads (4 waves), block tile 64x64, k-step 32.
// ---------------------------------------------------------------------------
__global__ __launch_bounds__(128)
void k_gemm_bf16(const unsigned short* __restrict__ A,
                 const unsigned short* __restrict__ B,
                 float* __restrict__ C, int M, int N, int K) {
    __shared__ alignas(16) unsigned short As[64 * 32];   // [row][k]
    __shared__ alignas(16) unsigned short Bs[64 * 32];   // [col][k]  (transposed)

    const int tid  = threadIdx.x;
    const int wv   = tid >> 5;
    const int lane = tid & 31;
    const int half = lane >> 4;
    const int lm   = lane & 15;
    const int m0   = blockIdx.y * 64;
    const int n0   = blockIdx.x * 64;
    const int rbase = (wv >> 1) * 32;
    const int cbase = (wv & 1) * 32;

    v8f acc[2][2] = {};

    for (int kk = 0; kk < K; kk += 32) {
        // stage A tile: 64 rows x 32 k
        {
            int row = tid >> 1, seg = (tid & 1) * 16;
            const us8* s0 = reinterpret_cast<const us8*>(A + (size_t)(m0 + row) * K + kk + seg);
            const us8* s1 = s0 + 1;
            us8* d0 = reinterpret_cast<us8*>(&As[row * 32 + seg]);
            d0[0] = *s0;
            d0[1] = *s1;
        }
        // stage B tile transposed: Bs[col][k]
        {
            int krow = tid >> 2, c8 = (tid & 3) * 8;
            us8 v = *reinterpret_cast<const us8*>(B + (size_t)(kk + krow) * N + n0 + c8);
#pragma unroll
            for (int jj = 0; jj < 8; ++jj)
                Bs[(c8 + jj) * 32 + krow] = v[jj];
        }
        if (kk + 32 < K) {
            __builtin_prefetch(A + (size_t)(m0 + (tid >> 1)) * K + kk + 32, 0, 0);
            __builtin_prefetch(B + (size_t)(kk + 32 + (tid >> 2)) * N + n0, 0, 0);
        }
        __syncthreads();

        v16bf a0 = load_frag(&As[(rbase + lm) * 32], 0, half);
        v16bf a1 = load_frag(&As[(rbase + 16 + lm) * 32], 0, half);
        v16bf b0 = load_frag(&Bs[(cbase + lm) * 32], 0, half);
        v16bf b1 = load_frag(&Bs[(cbase + 16 + lm) * 32], 0, half);

        acc[0][0] = wmma_bf16(a0, b0, acc[0][0]);
        acc[0][1] = wmma_bf16(a0, b1, acc[0][1]);
        acc[1][0] = wmma_bf16(a1, b0, acc[1][0]);
        acc[1][1] = wmma_bf16(a1, b1, acc[1][1]);
        __syncthreads();
    }

#pragma unroll
    for (int mi = 0; mi < 2; ++mi)
#pragma unroll
        for (int ni = 0; ni < 2; ++ni)
#pragma unroll
            for (int r = 0; r < 8; ++r) {
                int row = m0 + rbase + mi * 16 + r + 8 * half;
                int col = n0 + cbase + ni * 16 + lm;
                C[(size_t)row * N + col] = acc[mi][ni][r];
            }
}

// ---------------------------------------------------------------------------
// Flash attention with Transformer-XL relative shift.
// One wave per (b, head, 16-query tile). j-step = 32 keys.
// BD[i,j] = qr_i . rk[j - i + QLEN - 1]  (masked when j > i + mem_len)
// ---------------------------------------------------------------------------
__global__ __launch_bounds__(32)
void k_attn(const unsigned short* __restrict__ qw,   // [b][n][i][64]  q + r_w_bias
            const unsigned short* __restrict__ qr,   // [b][n][i][64]  q + r_r_bias
            const unsigned short* __restrict__ kb,   // [b][n][j][64]
            const unsigned short* __restrict__ vtb,  // [b][n][d][2048]
            const unsigned short* __restrict__ rkb,  // [n][p][64]
            float* __restrict__ av) {                // [i][b][n*64+d]
    __shared__ alignas(16) float          BDl[16 * 48];
    __shared__ alignas(16) unsigned short Pl[16 * 32];

    const int i0   = blockIdx.x * 16;
    const int h    = blockIdx.y;
    const int b    = blockIdx.z;
    const int lane = threadIdx.x;
    const int half = lane >> 4;
    const int lm   = lane & 15;

    const unsigned short* qwrow = qw + (((size_t)(b * NHEAD + h) * QLEN) + i0 + lm) * DHEAD;
    const unsigned short* qrrow = qr + (((size_t)(b * NHEAD + h) * QLEN) + i0 + lm) * DHEAD;
    v16bf qwf0 = load_frag(qwrow, 0, half),  qwf1 = load_frag(qwrow, 32, half);
    v16bf qrf0 = load_frag(qrrow, 0, half),  qrf1 = load_frag(qrrow, 32, half);

    const unsigned short* kbase = kb  + (size_t)(b * NHEAD + h) * KLEN * DHEAD;
    const unsigned short* vbase = vtb + (size_t)(b * NHEAD + h) * DHEAD * KLEN;
    const unsigned short* rbase = rkb + (size_t)h * KLEN * DHEAD;

    float m[8], l[8];
#pragma unroll
    for (int r = 0; r < 8; ++r) { m[r] = NEG_BIG; l[r] = 0.0f; }
    v8f o[4] = {};

    const int jlimit = min(KLEN, i0 + 16 + QLEN);  // beyond this everything is masked

    for (int j0 = 0; j0 < jlimit; j0 += 32) {
        // ---- AC = (q + r_w_bias) . k^T, two 16-col tiles ----
        v8f ac[2];
#pragma unroll
        for (int t = 0; t < 2; ++t) {
            const unsigned short* krow = kbase + (size_t)(j0 + 16 * t + lm) * DHEAD;
            v8f z = {};
            z = wmma_bf16(qwf0, load_frag(krow, 0, half), z);
            z = wmma_bf16(qwf1, load_frag(krow, 32, half), z);
            ac[t] = z;
        }
        // ---- BDraw over three d-sub-tiles covering didx in [0,48) ----
        const int dbase = j0 - i0 + (QLEN - 16);
#pragma unroll
        for (int s = 0; s < 3; ++s) {
            int p = dbase + 16 * s + lm;
            p = min(max(p, 0), KLEN - 1);
            const unsigned short* rrow = rbase + (size_t)p * DHEAD;
            v8f bd = {};
            bd = wmma_bf16(qrf0, load_frag(rrow, 0, half), bd);
            bd = wmma_bf16(qrf1, load_frag(rrow, 32, half), bd);
#pragma unroll
            for (int r = 0; r < 8; ++r)
                BDl[(r + 8 * half) * 48 + 16 * s + lm] = bd[r];
        }
        __syncthreads();

        // ---- scores + online softmax ----
        float sv[2][8];
#pragma unroll
        for (int t = 0; t < 2; ++t)
#pragma unroll
            for (int r = 0; r < 8; ++r) {
                int ti = r + 8 * half;
                int i  = i0 + ti;
                int jj = 16 * t + lm;
                int j  = j0 + jj;
                int didx = jj + 15 - ti;             // = (j-i+1023) - dbase
                float s = SCALE * (ac[t][r] + BDl[ti * 48 + didx]);
                sv[t][r] = (j > i + (KLEN - QLEN)) ? NEG_BIG : s;
            }

        float nm[8];
#pragma unroll
        for (int r = 0; r < 8; ++r) {
            float v = fmaxf(m[r], fmaxf(sv[0][r], sv[1][r]));
#pragma unroll
            for (int off = 1; off < 16; off <<= 1)
                v = fmaxf(v, __shfl_xor(v, off, 32));
            nm[r] = v;
        }

        float p0[8], p1[8], rs[8];
#pragma unroll
        for (int r = 0; r < 8; ++r) {
            float alpha = __expf(m[r] - nm[r]);
            m[r] = nm[r];
            p0[r] = __expf(sv[0][r] - nm[r]);
            p1[r] = __expf(sv[1][r] - nm[r]);
            float s = p0[r] + p1[r];
#pragma unroll
            for (int off = 1; off < 16; off <<= 1)
                s += __shfl_xor(s, off, 32);
            rs[r] = s;
            l[r] = l[r] * alpha + rs[r];
#pragma unroll
            for (int f = 0; f < 4; ++f)
                o[f][r] *= alpha;
        }

        // ---- stage P (bf16) to LDS and reload as an A-fragment ----
#pragma unroll
        for (int r = 0; r < 8; ++r) {
            int ti = r + 8 * half;
            Pl[ti * 32 + lm]      = f2bf(p0[r]);
            Pl[ti * 32 + 16 + lm] = f2bf(p1[r]);
        }
        __syncthreads();
        v16bf pf = load_frag(&Pl[lm * 32], 0, half);

        // ---- O += P @ V  (K = 32 keys, 4 d-tiles of 16) ----
#pragma unroll
        for (int f = 0; f < 4; ++f) {
            const unsigned short* vrow = vbase + (size_t)(f * 16 + lm) * KLEN;
            o[f] = wmma_bf16(pf, load_frag(vrow, j0, half), o[f]);
        }
    }

    // ---- finalize: attn_vec[i][b][n*64+d] = O / l ----
#pragma unroll
    for (int f = 0; f < 4; ++f)
#pragma unroll
        for (int r = 0; r < 8; ++r) {
            int i = i0 + r + 8 * half;
            float inv = 1.0f / l[r];
            av[((size_t)(i * BSZ + b) << 10) + (h << 6) + (f << 4) + lm] = o[f][r] * inv;
        }
}

// ---------------------------------------------------------------------------
// Residual + bias + LayerNorm. One 256-thread block per row of 1024.
// ---------------------------------------------------------------------------
__global__ __launch_bounds__(256)
void k_ln(const float* __restrict__ w, const float* __restrict__ ao,
          const float* __restrict__ bo, const float* __restrict__ g,
          const float* __restrict__ beta, float* __restrict__ out) {
    __shared__ float s1[256], s2[256];
    const int row = blockIdx.x;
    const int tid = threadIdx.x;
    float x[4], sum = 0.f, sq = 0.f;
#pragma unroll
    for (int q = 0; q < 4; ++q) {
        int c = tid + q * 256;
        float v = w[(size_t)row * DMODEL + c] + ao[(size_t)row * DMODEL + c] + bo[c];
        x[q] = v; sum += v; sq += v * v;
    }
    s1[tid] = sum; s2[tid] = sq;
    __syncthreads();
    for (int off = 128; off > 0; off >>= 1) {
        if (tid < off) { s1[tid] += s1[tid + off]; s2[tid] += s2[tid + off]; }
        __syncthreads();
    }
    float mu  = s1[0] * (1.0f / DMODEL);
    float var = s2[0] * (1.0f / DMODEL) - mu * mu;
    float rstd = rsqrtf(var + 1e-5f);
#pragma unroll
    for (int q = 0; q < 4; ++q) {
        int c = tid + q * 256;
        out[(size_t)row * DMODEL + c] = (x[q] - mu) * rstd * g[c] + beta[c];
    }
}

// ---------------------------------------------------------------------------
// Host launcher
// ---------------------------------------------------------------------------
extern "C" void kernel_launch(void* const* d_in, const int* in_sizes, int n_in,
                              void* d_out, int out_size, void* d_ws, size_t ws_size,
                              hipStream_t stream) {
    (void)in_sizes; (void)n_in; (void)out_size; (void)ws_size;
    const float* w    = (const float*)d_in[0];
    const float* r    = (const float*)d_in[1];
    const float* mems = (const float*)d_in[2];
    const float* rwb  = (const float*)d_in[3];
    const float* rrb  = (const float*)d_in[4];
    // d_in[5] = attn_mask (bool) -- mask applied analytically
    const float* Wq = (const float*)d_in[6];  const float* bq = (const float*)d_in[7];
    const float* Wk = (const float*)d_in[8];  const float* bk = (const float*)d_in[9];
    const float* Wv = (const float*)d_in[10]; const float* bv = (const float*)d_in[11];
    const float* Wr = (const float*)d_in[12]; const float* br = (const float*)d_in[13];
    const float* Wo = (const float*)d_in[14]; const float* bo = (const float*)d_in[15];
    const float* lng = (const float*)d_in[16]; const float* lnb = (const float*)d_in[17];
    float* out = (float*)d_out;

    char* base = (char*)d_ws;
    size_t off = 0;
    auto carve = [&](size_t bytes) { char* p = base + off; off = (off + bytes + 255) & ~(size_t)255; return p; };

    unsigned short* catb = (unsigned short*)carve((size_t)KLEN * BSZ * DMODEL * 2);
    unsigned short* rb   = (unsigned short*)carve((size_t)KLEN * DMODEL * 2);
    unsigned short* Wqb  = (unsigned short*)carve((size_t)DMODEL * DMODEL * 2);
    unsigned short* Wkb  = (unsigned short*)carve((size_t)DMODEL * DMODEL * 2);
    unsigned short* Wvb  = (unsigned short*)carve((size_t)DMODEL * DMODEL * 2);
    unsigned short* Wrb  = (unsigned short*)carve((size_t)DMODEL * DMODEL * 2);
    unsigned short* Wob  = (unsigned short*)carve((size_t)DMODEL * DMODEL * 2);
    float* qf  = (float*)carve((size_t)QLEN * BSZ * DMODEL * 4);   // reused as av
    float* kf  = (float*)carve((size_t)KLEN * BSZ * DMODEL * 4);   // reused as aof
    float* vf  = (float*)carve((size_t)KLEN * BSZ * DMODEL * 4);   // reused as avb
    float* rkf = (float*)carve((size_t)KLEN * DMODEL * 4);
    unsigned short* qwb  = (unsigned short*)carve((size_t)BSZ * NHEAD * QLEN * DHEAD * 2);
    unsigned short* qrb  = (unsigned short*)carve((size_t)BSZ * NHEAD * QLEN * DHEAD * 2);
    unsigned short* kbuf = (unsigned short*)carve((size_t)BSZ * NHEAD * KLEN * DHEAD * 2);
    unsigned short* vtb  = (unsigned short*)carve((size_t)BSZ * NHEAD * DHEAD * KLEN * 2);
    unsigned short* rkbf = (unsigned short*)carve((size_t)NHEAD * KLEN * DHEAD * 2);
    float* av            = qf;                    // alias (qf dead after k_make_q)
    float* aof           = kf;                    // alias (kf dead after k_make_k)
    unsigned short* avb  = (unsigned short*)vf;   // alias (vf dead after k_make_vt)

    const int TB = 256;
    // 1) bf16 conversions
    k_build_cat<<<4096, TB, 0, stream>>>(w, mems, catb);
    k_f2bf<<<2048, TB, 0, stream>>>(r,  rb,  KLEN * DMODEL);
    k_f2bf<<<1024, TB, 0, stream>>>(Wq, Wqb, DMODEL * DMODEL);
    k_f2bf<<<1024, TB, 0, stream>>>(Wk, Wkb, DMODEL * DMODEL);
    k_f2bf<<<1024, TB, 0, stream>>>(Wv, Wvb, DMODEL * DMODEL);
    k_f2bf<<<1024, TB, 0, stream>>>(Wr, Wrb, DMODEL * DMODEL);
    k_f2bf<<<1024, TB, 0, stream>>>(Wo, Wob, DMODEL * DMODEL);

    // 2) projections (q only needs the last qlen rows of cat)
    k_gemm_bf16<<<dim3(DMODEL / 64, (QLEN * BSZ) / 64), 128, 0, stream>>>(
        catb + (size_t)QLEN * BSZ * DMODEL, Wqb, qf, QLEN * BSZ, DMODEL, DMODEL);
    k_gemm_bf16<<<dim3(DMODEL / 64, (KLEN * BSZ) / 64), 128, 0, stream>>>(
        catb, Wkb, kf, KLEN * BSZ, DMODEL, DMODEL);
    k_gemm_bf16<<<dim3(DMODEL / 64, (KLEN * BSZ) / 64), 128, 0, stream>>>(
        catb, Wvb, vf, KLEN * BSZ, DMODEL, DMODEL);
    k_gemm_bf16<<<dim3(DMODEL / 64, KLEN / 64), 128, 0, stream>>>(
        rb, Wrb, rkf, KLEN, DMODEL, DMODEL);

    // 3) attention-layout epilogues
    k_make_q<<<2048, TB, 0, stream>>>(qf, bq, rwb, rrb, qwb, qrb);
    k_make_k<<<4096, TB, 0, stream>>>(kf, bk, kbuf);
    k_make_vt<<<4096, TB, 0, stream>>>(vf, bv, vtb);
    k_make_rk<<<2048, TB, 0, stream>>>(rkf, br, rkbf);

    // 4) flash attention with relative shift
    k_attn<<<dim3(QLEN / 16, NHEAD, BSZ), 32, 0, stream>>>(qwb, qrb, kbuf, vtb, rkbf, av);

    // 5) output projection + residual layernorm
    k_f2bf<<<2048, TB, 0, stream>>>(av, avb, QLEN * BSZ * DMODEL);
    k_gemm_bf16<<<dim3(DMODEL / 64, (QLEN * BSZ) / 64), 128, 0, stream>>>(
        avb, Wob, aof, QLEN * BSZ, DMODEL, DMODEL);
    k_ln<<<QLEN * BSZ, TB, 0, stream>>>(w, aof, bo, lng, lnb, out);
}
